// BatchGGNNEncoder_16063177687561
// MI455X (gfx1250) — matrix operand order repaired and checked by reading, not compiled, not run on hardware
//
#include <hip/hip_runtime.h>

// ---------------- problem constants ----------------
constexpr int Bn  = 16;      // batch
constexpr int Nn  = 2048;    // nodes
constexpr int En  = 16384;   // edges
constexpr int Fn  = 215;     // in features
constexpr int FnP = 224;     // F padded to K-step multiple (32)
constexpr int Hn  = 256;     // hidden
constexpr int Ln  = 3;       // layers
constexpr int Tn  = 8;       // edge types (0..T)

typedef __attribute__((ext_vector_type(16))) __bf16   v16bf;
typedef __attribute__((ext_vector_type(8)))  float    v8f;
typedef __attribute__((ext_vector_type(4)))  unsigned u32x4;
typedef __attribute__((ext_vector_type(2)))  unsigned u32x2;
typedef __attribute__((ext_vector_type(4)))  float    f32x4;

// ---------------- WMMA helper (CDNA5 wave32) ----------------
__device__ __forceinline__ v8f wmma_bf16(v16bf a, v16bf b, v8f c) {
  return __builtin_amdgcn_wmma_f32_16x16x32_bf16(
      false, a, false, b, (short)0, c, false, false);
}

union FragBF { v16bf v; u32x4 q[2]; };

// A fragment from an LDS 16x32 bf16 row-major tile (2x ds_load_b128).
__device__ __forceinline__ v16bf frag_a_lds(const __bf16* tile, int lane) {
  FragBF a;
  int row = lane & 15, hi = (lane >> 4) & 1;
  const u32x4* p = (const u32x4*)tile + row * 4;
  a.q[0] = p[2 * hi];
  a.q[1] = p[2 * hi + 1];
  return a.v;
}

// A fragment straight from a row-major bf16 row in global memory.
// rowp must already include  row*stride + hi*16 + kb.
__device__ __forceinline__ v16bf frag_a_gl(const __bf16* rowp) {
  FragBF a;
  a.q[0] = *(const u32x4*)(rowp);        // K = kb + hi*16 .. +7
  a.q[1] = *(const u32x4*)(rowp + 8);    // K = kb + hi*16 + 8 .. +15
  return a.v;
}

// B fragment from transposed bf16 weights WT[col][K] (2x global b128).
__device__ __forceinline__ v16bf frag_b_gl(const __bf16* colp) {
  FragBF b;
  b.q[0] = ((const u32x4*)colp)[0];
  b.q[1] = ((const u32x4*)colp)[1];
  return b.v;
}

// Direct emission of the packed convert: dword = { bf16(a), bf16(b) }.
// (Mnemonic verified present in gfx1250 compiler output.)
__device__ __forceinline__ unsigned pack2bf(float a, float b) {
  unsigned r;
  asm("v_cvt_pk_bf16_f32 %0, %1, %2" : "=v"(r) : "v"(a), "v"(b));
  return r;
}

__device__ __forceinline__ float sigmoidf_fast(float x) {
  return 1.0f / (1.0f + __expf(-x));
}

// ------------- weight pre-pack: WT[n][k] = (bf16)W[k][n], K padded -------------
__global__ void ggnn_packT(const float* __restrict__ W, __bf16* __restrict__ WT,
                           int K, int KP, int NC, int total) {
  int idx = blockIdx.x * blockDim.x + threadIdx.x;
  if (idx >= total) return;
  int k = idx % KP;
  int n = idx / KP;
  float v = 0.0f;
  if (k < K) v = W[(long)k * NC + n];
  WT[idx] = (__bf16)v;
}

// ---------------- has_edges per batch ----------------
__global__ void ggnn_has_edges(const int* __restrict__ ei,
                               const int* __restrict__ nn,
                               int* __restrict__ hasE) {
  __shared__ int flag;
  int b = blockIdx.x;
  if (threadIdx.x == 0) flag = 0;
  __syncthreads();
  int num = nn[b];
  int any = 0;
  for (int e = threadIdx.x; e < En; e += blockDim.x) {
    int s = ei[(long)b * 2 * En + e];
    int d = ei[(long)b * 2 * En + En + e];
    any |= (s < num) && (d < num);
  }
  if (any) atomicOr(&flag, 1);
  __syncthreads();
  if (threadIdx.x == 0) hasE[b] = flag;
}

// ---------------- projection: h = NF @ Wp + bp ----------------
// grid.x = B*N/16, grid.y = 2; block = 128; each wave: two 16-col tiles.
// LDS A-tile double-buffered: one barrier per K-step.
__global__ void ggnn_project(const float* __restrict__ nf,
                             const __bf16* __restrict__ WpT,
                             const float* __restrict__ bp,
                             float* __restrict__ hf,
                             __bf16* __restrict__ hb) {
  __shared__ __bf16 sA[2][16 * 32];
  const int lane = threadIdx.x & 31;
  const int wave = threadIdx.x >> 5;
  const int hi   = lane >> 4;
  const int c0   = (blockIdx.y * 8 + wave * 2) * 16;
  const long rbase = (long)blockIdx.x * 16;

  const __bf16* bptr0 = WpT + (long)(c0 + (lane & 15)) * FnP + hi * 16;
  const __bf16* bptr1 = bptr0 + 16 * FnP;
  const int idx4 = threadIdx.x * 4;           // 128 threads * 4 = 512 elems
  const int am = idx4 >> 5, ak = idx4 & 31;
  const float* arow = nf + (rbase + am) * Fn + ak;

  auto fill = [&](int buf, int kb) {
    float x0 = 0.f, x1 = 0.f, x2 = 0.f, x3 = 0.f;
    int gk = kb + ak;
    if (gk + 0 < Fn) x0 = arow[kb + 0];
    if (gk + 1 < Fn) x1 = arow[kb + 1];
    if (gk + 2 < Fn) x2 = arow[kb + 2];
    if (gk + 3 < Fn) x3 = arow[kb + 3];
    u32x2 pk; pk.x = pack2bf(x0, x1); pk.y = pack2bf(x2, x3);
    ((u32x2*)sA[buf])[threadIdx.x] = pk;      // one ds_store_b64
  };

  fill(0, 0);
  v8f acc0 = {}, acc1 = {};
  constexpr int NIT = FnP / 32;               // 7
  for (int i = 0; i < NIT; ++i) {
    int kb = i * 32;
    __syncthreads();
    if (i + 1 < NIT) fill((i + 1) & 1, kb + 32);
    v16bf af = frag_a_lds(sA[i & 1], lane);
    acc0 = wmma_bf16(af, frag_b_gl(bptr0 + kb), acc0);
    acc1 = wmma_bf16(af, frag_b_gl(bptr1 + kb), acc1);
  }

  int col0 = c0 + (lane & 15);
  float bias0 = bp[col0], bias1 = bp[col0 + 16];
#pragma unroll
  for (int r = 0; r < 8; ++r) {
    long row = rbase + r + hi * 8;
    float v0 = acc0[r] + bias0;
    float v1 = acc1[r] + bias1;
    hf[row * Hn + col0]      = v0;
    hf[row * Hn + col0 + 16] = v1;
    hb[row * Hn + col0]      = (__bf16)v0;
    hb[row * Hn + col0 + 16] = (__bf16)v1;
  }
}

// ---------------- edge messages + f32 atomic scatter ----------------
// grid.x = B * E/16, grid.y = 2; block = 128; each wave: two 16-col tiles.
// Gathered A fragment loaded once per K-step, feeds both WMMAs.
__global__ void ggnn_messages(const __bf16* __restrict__ hb,
                              const int* __restrict__ ei,
                              const int* __restrict__ etyp,
                              const int* __restrict__ nn,
                              const __bf16* __restrict__ WmT,
                              const float* __restrict__ bm,
                              const float* __restrict__ Eemb,
                              float* __restrict__ msg, int l) {
  __shared__ int sSrc[16], sDst[16], sEt[16], sVal[16];
  const int lane = threadIdx.x & 31;
  const int wave = threadIdx.x >> 5;
  const int hi   = lane >> 4;
  const int ET = En / 16;
  const int b = blockIdx.x / ET;
  const int etile = blockIdx.x % ET;
  const int c0 = (blockIdx.y * 8 + wave * 2) * 16;

  if (threadIdx.x < 16) {
    int e = etile * 16 + threadIdx.x;
    int s = ei[(long)b * 2 * En + e];
    int d = ei[(long)b * 2 * En + En + e];
    int t = etyp[(long)b * En + e];
    t = t < 0 ? 0 : (t > Tn ? Tn : t);
    int num = nn[b];
    sVal[threadIdx.x] = (s < num) && (d < num);
    sSrc[threadIdx.x] = ((unsigned)s < (unsigned)Nn) ? s : 0;
    sDst[threadIdx.x] = ((unsigned)d < (unsigned)Nn) ? d : 0;
    sEt[threadIdx.x]  = t;
  }
  __syncthreads();

  const __bf16* aptr  = hb + ((long)b * Nn + sSrc[lane & 15]) * Hn + hi * 16;
  const __bf16* bptr0 = WmT + ((long)l * Hn + c0 + (lane & 15)) * Hn + hi * 16;
  const __bf16* bptr1 = bptr0 + 16 * Hn;

  v8f acc0 = {}, acc1 = {};
#pragma unroll
  for (int kb = 0; kb < Hn; kb += 32) {
    v16bf af = frag_a_gl(aptr + kb);
    acc0 = wmma_bf16(af, frag_b_gl(bptr0 + kb), acc0);
    acc1 = wmma_bf16(af, frag_b_gl(bptr1 + kb), acc1);
  }

  int col0 = c0 + (lane & 15);
  float bmv0 = bm[(long)l * Hn + col0];
  float bmv1 = bm[(long)l * Hn + col0 + 16];
#pragma unroll
  for (int r = 0; r < 8; ++r) {
    int m = r + hi * 8;
    if (sVal[m]) {
      long ebase = ((long)l * (Tn + 1) + sEt[m]) * Hn;
      long dbase = ((long)b * Nn + sDst[m]) * Hn;
      float v0 = acc0[r] + bmv0 + Eemb[ebase + col0];
      float v1 = acc1[r] + bmv1 + Eemb[ebase + col0 + 16];
      __hip_atomic_fetch_add(&msg[dbase + col0], v0,
                             __ATOMIC_RELAXED, __HIP_MEMORY_SCOPE_AGENT);
      __hip_atomic_fetch_add(&msg[dbase + col0 + 16], v1,
                             __ATOMIC_RELAXED, __HIP_MEMORY_SCOPE_AGENT);
    }
  }
}

// ---------------- fused gate GEMMs + GRU update ----------------
// grid.x = B*N/16, grid.y = 4; block = 128. A_msg staged via double-buffered
// LDS (packed bf16 converts); A_h + all 6 B fragments direct global b128.
__global__ void ggnn_gru(const float* __restrict__ msg,
                         const float* __restrict__ hf,
                         const __bf16* __restrict__ hb,
                         const __bf16* __restrict__ WihT,
                         const __bf16* __restrict__ WhhT,
                         const float* __restrict__ bih,
                         const float* __restrict__ bhh,
                         const int* __restrict__ nn,
                         const int* __restrict__ hasE,
                         float* __restrict__ hfo,
                         __bf16* __restrict__ hbo, int l) {
  __shared__ __bf16 sAm[2][16 * 32];
  const int lane = threadIdx.x & 31;
  const int wave = threadIdx.x >> 5;
  const int hi   = lane >> 4;
  const int c0   = (blockIdx.y * 4 + wave) * 16;
  const long rbase = (long)blockIdx.x * 16;

  const __bf16* ah = hb + (rbase + (lane & 15)) * Hn + hi * 16;
  const long wl = (long)l * 3 * Hn * Hn;
  const int ncol = c0 + (lane & 15);
  const __bf16* bir_p = WihT + wl + (long)(0 * Hn + ncol) * Hn + hi * 16;
  const __bf16* biz_p = WihT + wl + (long)(1 * Hn + ncol) * Hn + hi * 16;
  const __bf16* bic_p = WihT + wl + (long)(2 * Hn + ncol) * Hn + hi * 16;
  const __bf16* bhr_p = WhhT + wl + (long)(0 * Hn + ncol) * Hn + hi * 16;
  const __bf16* bhz_p = WhhT + wl + (long)(1 * Hn + ncol) * Hn + hi * 16;
  const __bf16* bhc_p = WhhT + wl + (long)(2 * Hn + ncol) * Hn + hi * 16;

  const int idx4 = threadIdx.x * 4;
  const int am = idx4 >> 5, ak = idx4 & 31;
  const float* mrow = msg + (rbase + am) * Hn + ak;

  auto fill = [&](int buf, int kb) {
    f32x4 x = *(const f32x4*)(mrow + kb);     // 16B aligned
    u32x2 pk; pk.x = pack2bf(x.x, x.y); pk.y = pack2bf(x.z, x.w);
    ((u32x2*)sAm[buf])[threadIdx.x] = pk;
  };

  fill(0, 0);
  v8f ar = {}, az = {}, gic = {}, ghc = {};
  constexpr int NIT = Hn / 32;                // 8
  for (int i = 0; i < NIT; ++i) {
    int kb = i * 32;
    __syncthreads();
    if (i + 1 < NIT) fill((i + 1) & 1, kb + 32);
    v16bf amf = frag_a_lds(sAm[i & 1], lane);
    v16bf ahf = frag_a_gl(ah + kb);
    ar  = wmma_bf16(amf, frag_b_gl(bir_p + kb), ar);
    ar  = wmma_bf16(ahf, frag_b_gl(bhr_p + kb), ar);
    az  = wmma_bf16(amf, frag_b_gl(biz_p + kb), az);
    az  = wmma_bf16(ahf, frag_b_gl(bhz_p + kb), az);
    gic = wmma_bf16(amf, frag_b_gl(bic_p + kb), gic);
    ghc = wmma_bf16(ahf, frag_b_gl(bhc_p + kb), ghc);
  }

  const long bb = (long)l * 3 * Hn;
  float br0 = bih[bb + ncol]      + bhh[bb + ncol];
  float bz0 = bih[bb + Hn + ncol] + bhh[bb + Hn + ncol];
  float bic = bih[bb + 2 * Hn + ncol];
  float bhc = bhh[bb + 2 * Hn + ncol];
#pragma unroll
  for (int r = 0; r < 8; ++r) {
    long grow = rbase + r + hi * 8;
    int bidx = (int)(grow >> 11);        // / N
    int nidx = (int)(grow & (Nn - 1));   // % N
    float hold = hf[grow * Hn + ncol];
    float rg = sigmoidf_fast(ar[r] + br0);
    float zg = sigmoidf_fast(az[r] + bz0);
    float cg = tanhf(gic[r] + bic + rg * (ghc[r] + bhc));
    float hnew = (1.0f - zg) * cg + zg * hold;
    bool upd = (nidx < nn[bidx]) && (hasE[bidx] != 0);
    float v = upd ? hnew : hold;
    hfo[grow * Hn + ncol] = v;
    hbo[grow * Hn + ncol] = (__bf16)v;
  }
}

// ---------------- masked node sum -> (B, H) ----------------
// block = 64 threads, 4 cols each (b128 loads/stores).
__global__ void ggnn_reduce(const float* __restrict__ hf,
                            const int* __restrict__ nn,
                            float* __restrict__ out) {
  int b = blockIdx.x;
  int col4 = threadIdx.x * 4;
  int num = nn[b];
  f32x4 acc = {};
  const float* base = hf + (long)b * Nn * Hn + col4;
  for (int n = 0; n < num; ++n) {
    f32x4 x = *(const f32x4*)(base + (long)n * Hn);
    acc += x;
  }
  *(f32x4*)(out + b * Hn + col4) = acc;
}

// ---------------- host launcher ----------------
extern "C" void kernel_launch(void* const* d_in, const int* in_sizes, int n_in,
                              void* d_out, int out_size, void* d_ws, size_t ws_size,
                              hipStream_t stream) {
  (void)in_sizes; (void)n_in; (void)out_size; (void)ws_size;
  const float* nf  = (const float*)d_in[0];
  const int*   ei  = (const int*)d_in[1];
  const int*   etp = (const int*)d_in[2];
  const int*   nnp = (const int*)d_in[3];
  const float* Wp  = (const float*)d_in[4];
  const float* bp  = (const float*)d_in[5];
  const float* Wm  = (const float*)d_in[6];
  const float* bm  = (const float*)d_in[7];
  const float* Ee  = (const float*)d_in[8];
  const float* Wih = (const float*)d_in[9];
  const float* Whh = (const float*)d_in[10];
  const float* bih = (const float*)d_in[11];
  const float* bhh = (const float*)d_in[12];
  float* out = (float*)d_out;

  char* p = (char*)d_ws;
  const size_t szHf = (size_t)Bn * Nn * Hn * sizeof(float);   // 32 MB
  const size_t szHb = (size_t)Bn * Nn * Hn * sizeof(__bf16);  // 16 MB
  float*  hAf = (float*)p;  p += szHf;
  float*  hBf = (float*)p;  p += szHf;
  __bf16* hAb = (__bf16*)p; p += szHb;
  __bf16* hBb = (__bf16*)p; p += szHb;
  float*  msg = (float*)p;  p += szHf;
  int*    hasE = (int*)p;   p += 256;
  __bf16* WpT  = (__bf16*)p; p += (size_t)Hn * FnP * 2;           // [256][224]
  __bf16* WmT  = (__bf16*)p; p += (size_t)Ln * Hn * Hn * 2;       // [l][256][256]
  __bf16* WihT = (__bf16*)p; p += (size_t)Ln * 3 * Hn * Hn * 2;   // [l][768][256]
  __bf16* WhhT = (__bf16*)p; p += (size_t)Ln * 3 * Hn * Hn * 2;

  // ---- weight pre-pack (bf16, transposed [col][K]) ----
  {
    int tot = Hn * FnP;
    ggnn_packT<<<(tot + 255) / 256, 256, 0, stream>>>(Wp, WpT, Fn, FnP, Hn, tot);
    for (int l = 0; l < Ln; ++l) {
      int t1 = Hn * Hn;
      ggnn_packT<<<(t1 + 255) / 256, 256, 0, stream>>>(
          Wm + (size_t)l * Hn * Hn, WmT + (size_t)l * Hn * Hn, Hn, Hn, Hn, t1);
      int t2 = 3 * Hn * Hn;
      ggnn_packT<<<(t2 + 255) / 256, 256, 0, stream>>>(
          Wih + (size_t)l * Hn * 3 * Hn, WihT + (size_t)l * 3 * Hn * Hn,
          Hn, Hn, 3 * Hn, t2);
      ggnn_packT<<<(t2 + 255) / 256, 256, 0, stream>>>(
          Whh + (size_t)l * Hn * 3 * Hn, WhhT + (size_t)l * 3 * Hn * Hn,
          Hn, Hn, 3 * Hn, t2);
    }
  }

  ggnn_has_edges<<<Bn, 256, 0, stream>>>(ei, nnp, hasE);
  ggnn_project<<<dim3(Bn * Nn / 16, 2), 128, 0, stream>>>(nf, WpT, bp, hAf, hAb);

  float*  hcf = hAf; float*  hnf = hBf;
  __bf16* hcb = hAb; __bf16* hnb = hBb;
  for (int l = 0; l < Ln; ++l) {
    hipMemsetAsync(msg, 0, szHf, stream);
    ggnn_messages<<<dim3(Bn * (En / 16), 2), 128, 0, stream>>>(
        hcb, ei, etp, nnp, WmT, bm, Ee, msg, l);
    ggnn_gru<<<dim3(Bn * Nn / 16, 4), 128, 0, stream>>>(
        msg, hcf, hcb, WihT, WhhT, bih, bhh, nnp, hasE, hnf, hnb, l);
    float* tf = hcf; hcf = hnf; hnf = tf;
    __bf16* tb = hcb; hcb = hnb; hnb = tb;
  }
  ggnn_reduce<<<Bn, 64, 0, stream>>>(hcf, nnp, out);
}
